// CrossAttentionFusion_83270825934975
// MI455X (gfx1250) — matrix operand order repaired and checked by reading, not compiled, not run on hardware
//
#include <hip/hip_runtime.h>
#include <hip/hip_bf16.h>
#include <math.h>
#include <stdint.h>

typedef __bf16 bf16_t;
typedef __attribute__((ext_vector_type(16))) __bf16 v16bf;
typedef __attribute__((ext_vector_type(8)))  __bf16 v8bf;
typedef __attribute__((ext_vector_type(4)))  __bf16 v4bf;
typedef __attribute__((ext_vector_type(8)))  float  v8f;

// ---------------------------------------------------------------------------
// fp32 -> bf16 conversion, 4 elements/thread (all sizes are multiples of 4)
// ---------------------------------------------------------------------------
__global__ __launch_bounds__(256)
void cvt_kernel(const float* __restrict__ src, bf16_t* __restrict__ dst, int n4) {
    int i = blockIdx.x * blockDim.x + threadIdx.x;
    if (i < n4) {
        const float4 f = ((const float4*)src)[i];
        v4bf o;
        o[0] = (bf16_t)f.x; o[1] = (bf16_t)f.y;
        o[2] = (bf16_t)f.z; o[3] = (bf16_t)f.w;
        ((v4bf*)dst)[i] = o;
    }
}

// ---------------------------------------------------------------------------
// WMMA GEMM:  C[M,N] = A[M,K] (row-major bf16) @ W[N,K]^T (row-major bf16) + bias
// EPI == 0 : store bf16 to Cb (leading dim ldc)
// EPI == 1 : exact-erf GELU, store fp32 to Cf (leading dim ldc)
// Block: 128x128 tile, BK=64, 256 threads (8 wave32), wave tile 32x64.
// Double-buffered LDS fed by GLOBAL_LOAD_ASYNC_TO_LDS_B128 (ASYNCcnt pipeline).
// M,N multiples of 128; K multiple of 64 (true for all our shapes).
// ---------------------------------------------------------------------------
template<int EPI>
__global__ __launch_bounds__(256, 2)
void wmma_gemm_kernel(const bf16_t* __restrict__ A, const bf16_t* __restrict__ W,
                      const float* __restrict__ bias,
                      bf16_t* __restrict__ Cb, float* __restrict__ Cf,
                      int M, int N, int K, int ldc) {
    // 2 stages x (A tile 128x64 bf16 = 16KB, B tile 128x64 bf16 = 16KB) = 64KB
    __shared__ __align__(128) bf16_t smem[2 * 16384];

    const int tid   = threadIdx.x;
    const int lane  = tid & 31;
    const int wave  = tid >> 5;          // 0..7
    const int wm    = (wave >> 1) * 32;  // wave M-offset in block tile
    const int wn    = (wave & 1) * 64;   // wave N-offset in block tile
    const int bm0   = blockIdx.x * 128;
    const int bn0   = blockIdx.y * 128;
    const int lrow  = lane & 15;
    const int lhalf = lane >> 4;         // 0 or 1

    // Low 32 bits of the generic LDS address == LDS byte offset (aperture base
    // has zero low bits per ISA flat addressing).
    const uint32_t ldsBase = (uint32_t)(uintptr_t)(void*)smem;

    // Per-thread staging geometry: 4 x 16B chunks for A and 4 for B per stage.
    uint32_t ldsOff[4];   // byte offset within a 128x64 bf16 tile
    uint32_t gOff[4];     // byte offset within source matrix row-block (no k0)
    #pragma unroll
    for (int i = 0; i < 4; ++i) {
        const int idx = tid + i * 256;   // 0..1023
        const int r   = idx >> 3;        // 0..127
        const int c8  = (idx & 7) << 3;  // 0..56 step 8 (bf16 elems)
        ldsOff[i] = (uint32_t)((r * 64 + c8) * 2);
        gOff[i]   = (uint32_t)((r * K + c8) * 2);
    }
    const uint64_t aBase = (uint64_t)(uintptr_t)(A + (size_t)bm0 * K);
    const uint64_t bBase = (uint64_t)(uintptr_t)(W + (size_t)bn0 * K);

    // Issue one stage (8 async-to-LDS b128 loads) into buffer `buf`.
    auto issue_stage = [&](int k0, int buf) {
        const uint32_t bufBase = ldsBase + (uint32_t)buf * 32768u;
        const uint32_t kByte   = (uint32_t)(k0 * 2);
        #pragma unroll
        for (int i = 0; i < 4; ++i) {
            const uint32_t la = bufBase + ldsOff[i];
            const uint32_t va = gOff[i] + kByte;
            asm volatile("global_load_async_to_lds_b128 %0, %1, %2"
                         :: "v"(la), "v"(va), "s"(aBase) : "memory");
            const uint32_t lb = bufBase + 16384u + ldsOff[i];
            asm volatile("global_load_async_to_lds_b128 %0, %1, %2"
                         :: "v"(lb), "v"(va), "s"(bBase) : "memory");
        }
    };

    v8f acc[2][4] = {};

    issue_stage(0, 0);
    if (K > 64) issue_stage(64, 1);

    for (int k0 = 0, stage = 0; k0 < K; k0 += 64, ++stage) {
        // Stage k landed when <= 8 async ops (stage k+1's) remain outstanding.
        if (k0 + 64 < K)
            asm volatile("s_wait_asynccnt 0x8" ::: "memory");
        else
            asm volatile("s_wait_asynccnt 0x0" ::: "memory");
        __syncthreads();

        const bf16_t* sA = smem + (stage & 1) * 16384;
        const bf16_t* sB = sA + 8192;

        #pragma unroll
        for (int kk = 0; kk < 64; kk += 32) {
            // A fragments (ISA 7.12.2, 16-bit A 16x32):
            // lane<16: elems 0..7 = K[kb..kb+7], elems 8..15 = K[kb+16..kb+23], kb=0
            // lane>=16: same with kb=8
            v16bf af[2];
            #pragma unroll
            for (int i = 0; i < 2; ++i) {
                const int row = wm + i * 16 + lrow;
                const int kb  = kk + lhalf * 8;
                v8bf lo = *(const v8bf*)(&sA[row * 64 + kb]);
                v8bf hi = *(const v8bf*)(&sA[row * 64 + kb + 16]);
                #pragma unroll
                for (int e = 0; e < 8; ++e) { af[i][e] = lo[e]; af[i][e + 8] = hi[e]; }
            }
            // B fragments (16-bit B 32x16): lane = column N, same K order as A:
            // elems 0..7 = K[kb..kb+7], elems 8..15 = K[kb+16..kb+23], kb = 8*(lane>=16).
            v16bf bfrag[4];
            #pragma unroll
            for (int j = 0; j < 4; ++j) {
                const int row = wn + j * 16 + lrow;   // weight row == output col
                const int kb  = kk + lhalf * 8;
                v8bf lo = *(const v8bf*)(&sB[row * 64 + kb]);
                v8bf hi = *(const v8bf*)(&sB[row * 64 + kb + 16]);
                #pragma unroll
                for (int e = 0; e < 8; ++e) { bfrag[j][e] = lo[e]; bfrag[j][e + 8] = hi[e]; }
            }
            #pragma unroll
            for (int i = 0; i < 2; ++i)
                #pragma unroll
                for (int j = 0; j < 4; ++j)
                    acc[i][j] = __builtin_amdgcn_wmma_f32_16x16x32_bf16(
                        false, af[i], false, bfrag[j], (short)0, acc[i][j],
                        false, false);
        }
        __syncthreads();  // all waves done reading buf[stage&1]
        if (k0 + 128 < K) issue_stage(k0 + 128, stage & 1);
    }

    // ---- epilogue: C/D layout = VGPR r -> row r (lanes 0..15) / row 8+r ----
    #pragma unroll
    for (int i = 0; i < 2; ++i) {
        #pragma unroll
        for (int j = 0; j < 4; ++j) {
            const int col = bn0 + wn + j * 16 + lrow;
            const float bv = bias[col];
            #pragma unroll
            for (int r = 0; r < 8; ++r) {
                const int row = bm0 + wm + i * 16 + lhalf * 8 + r;
                float v = acc[i][j][r] + bv;
                if constexpr (EPI == 1) {
                    v = 0.5f * v * (1.0f + erff(v * 0.70710678118654752f));
                    Cf[(size_t)row * ldc + col] = v;
                } else {
                    Cb[(size_t)row * ldc + col] = (bf16_t)v;
                }
            }
        }
    }
}

// ---------------------------------------------------------------------------
// In-place LayerNorm over rows of 2048 bf16 (one block per row, 8 elems/thread)
// ---------------------------------------------------------------------------
__global__ __launch_bounds__(256)
void ln_kernel(bf16_t* __restrict__ Z, const float* __restrict__ g,
               const float* __restrict__ b) {
    const int row = blockIdx.x;
    const int tid = threadIdx.x;
    bf16_t* zr = Z + (size_t)row * 2048;

    v8bf v = *(const v8bf*)(&zr[tid * 8]);
    float x[8];
    float s = 0.f, s2 = 0.f;
    #pragma unroll
    for (int e = 0; e < 8; ++e) {
        x[e] = (float)v[e];
        s += x[e];
        s2 += x[e] * x[e];
    }
    __shared__ float rs[256];
    __shared__ float rq[256];
    rs[tid] = s; rq[tid] = s2;
    __syncthreads();
    for (int o = 128; o > 0; o >>= 1) {
        if (tid < o) { rs[tid] += rs[tid + o]; rq[tid] += rq[tid + o]; }
        __syncthreads();
    }
    const float mean = rs[0] * (1.0f / 2048.0f);
    const float var  = rq[0] * (1.0f / 2048.0f) - mean * mean;
    const float inv  = rsqrtf(var + 1e-5f);

    v8bf o8;
    #pragma unroll
    for (int e = 0; e < 8; ++e) {
        const int c = tid * 8 + e;
        o8[e] = (bf16_t)((x[e] - mean) * inv * g[c] + b[c]);
    }
    *(v8bf*)(&zr[tid * 8]) = o8;
}

// ---------------------------------------------------------------------------
extern "C" void kernel_launch(void* const* d_in, const int* in_sizes, int n_in,
                              void* d_out, int out_size, void* d_ws, size_t ws_size,
                              hipStream_t stream) {
    (void)in_sizes; (void)n_in; (void)out_size; (void)ws_size;
    constexpr int Bn = 16384, D = 1024;

    const float* x_u    = (const float*)d_in[0];
    const float* x_m    = (const float*)d_in[1];
    const float* w_qkv1 = (const float*)d_in[2];
    const float* b_qkv1 = (const float*)d_in[3];
    const float* w_o1   = (const float*)d_in[4];
    const float* b_o1   = (const float*)d_in[5];
    const float* w_qkv2 = (const float*)d_in[6];
    const float* b_qkv2 = (const float*)d_in[7];
    const float* w_o2   = (const float*)d_in[8];
    const float* b_o2   = (const float*)d_in[9];
    const float* ln_g   = (const float*)d_in[10];
    const float* ln_b   = (const float*)d_in[11];
    const float* w_proj = (const float*)d_in[12];
    const float* b_proj = (const float*)d_in[13];
    float* out = (float*)d_out;

    char* ws = (char*)d_ws;
    size_t off = 0;
    auto take = [&](size_t bytes) -> void* {
        void* p = ws + off;
        off += (bytes + 255) & ~(size_t)255;
        return p;
    };
    bf16_t* Xu  = (bf16_t*)take((size_t)Bn * D * 2);      // 32 MB
    bf16_t* Xm  = (bf16_t*)take((size_t)Bn * D * 2);      // 32 MB
    bf16_t* Wv1 = (bf16_t*)take((size_t)D * D * 2);       //  2 MB
    bf16_t* Wo1 = (bf16_t*)take((size_t)D * D * 2);
    bf16_t* Wv2 = (bf16_t*)take((size_t)D * D * 2);
    bf16_t* Wo2 = (bf16_t*)take((size_t)D * D * 2);
    bf16_t* Wp  = (bf16_t*)take((size_t)2 * D * D * 2);   //  4 MB
    bf16_t* V   = (bf16_t*)take((size_t)Bn * D * 2);      // 32 MB (reused)
    bf16_t* Z   = (bf16_t*)take((size_t)Bn * 2 * D * 2);  // 64 MB

    auto cvt = [&](const float* s, bf16_t* dp, size_t n) {
        const int n4 = (int)(n / 4);
        cvt_kernel<<<(n4 + 255) / 256, 256, 0, stream>>>(s, dp, n4);
    };
    cvt(x_u, Xu, (size_t)Bn * D);
    cvt(x_m, Xm, (size_t)Bn * D);
    cvt(w_qkv1 + (size_t)2 * D * D, Wv1, (size_t)D * D);  // V-block of packed in_proj
    cvt(w_o1, Wo1, (size_t)D * D);
    cvt(w_qkv2 + (size_t)2 * D * D, Wv2, (size_t)D * D);
    cvt(w_o2, Wo2, (size_t)D * D);
    cvt(w_proj, Wp, (size_t)2 * D * D);

    const dim3 blk(256);
    const dim3 grd(Bn / 128, D / 128);

    // softmax over a single key == 1  =>  attention output == V projection.
    // Path 1: z1 = (x_u Wv1^T + bv1) Wo1^T + bo1  -> Z[:, 0:D]
    wmma_gemm_kernel<0><<<grd, blk, 0, stream>>>(Xu, Wv1, b_qkv1 + 2 * D, V, nullptr,
                                                 Bn, D, D, D);
    wmma_gemm_kernel<0><<<grd, blk, 0, stream>>>(V, Wo1, b_o1, Z, nullptr,
                                                 Bn, D, D, 2 * D);
    // Path 2: z2 = (x_m Wv2^T + bv2) Wo2^T + bo2  -> Z[:, D:2D]
    wmma_gemm_kernel<0><<<grd, blk, 0, stream>>>(Xm, Wv2, b_qkv2 + 2 * D, V, nullptr,
                                                 Bn, D, D, D);
    wmma_gemm_kernel<0><<<grd, blk, 0, stream>>>(V, Wo2, b_o2, Z + D, nullptr,
                                                 Bn, D, D, 2 * D);
    // LayerNorm(2048) in place on Z
    ln_kernel<<<Bn, 256, 0, stream>>>(Z, ln_g, ln_b);
    // h = GELU(Zn Wproj^T + bproj) -> fp32 output
    wmma_gemm_kernel<1><<<grd, blk, 0, stream>>>(Z, Wp, b_proj, nullptr, out,
                                                 Bn, D, 2 * D, D);
}
